// CausalSelfAttention_40415642256015
// MI455X (gfx1250) — matrix operand order repaired and checked by reading, not compiled, not run on hardware
//
#include <hip/hip_runtime.h>
#include <hip/hip_bf16.h>

// CDNA5 / gfx1250 causal self-attention (B=2, N=2048, E=1024, H=16, Kd=64).
// All matmuls via v_wmma_f32_16x16x32_f16 (wave32).
// Faithful to reference: scale = 1/sqrt(H) = 0.25, mask where query>key,
// softmax normalizes over the QUERY axis (per-key stats) -> two-pass attention.
// GEMMs: double-buffered LDS staging; async global->LDS copies when the
// toolchain exposes the gfx1250 async builtins (guarded by __has_builtin).

typedef __attribute__((ext_vector_type(16))) _Float16 v16h;
typedef __attribute__((ext_vector_type(8)))  float    v8f;
typedef __attribute__((ext_vector_type(4)))  _Float16 v4h;
typedef __attribute__((ext_vector_type(4)))  int      v4i;

union AFrag { v16h v; uint4 u[2]; };          // 32 bytes: A or B operand fragment
union CFrag { v8f  v; float f[8]; };          // 32 bytes: C/D accumulator fragment

#define WMMA16x16x32(a, b, c) \
  __builtin_amdgcn_wmma_f32_16x16x32_f16(false, (a), false, (b), (short)0, (c), false, false)

#if __has_builtin(__builtin_amdgcn_global_load_async_to_lds_b128) && \
    __has_builtin(__builtin_amdgcn_s_wait_asynccnt)
#define HAVE_ASYNC_LDS 1
#else
#define HAVE_ASYNC_LDS 0
#endif

// Copy 32 bytes (16 halves) from global to LDS, 16B-aligned on both sides.
__device__ __forceinline__ void copy32_to_lds(const _Float16* g, _Float16* l) {
#if HAVE_ASYNC_LDS
  __builtin_amdgcn_global_load_async_to_lds_b128(
      (__attribute__((address_space(1))) v4i*)g,
      (__attribute__((address_space(3))) v4i*)l, 0, 0);
  __builtin_amdgcn_global_load_async_to_lds_b128(
      (__attribute__((address_space(1))) v4i*)(g + 8),
      (__attribute__((address_space(3))) v4i*)(l + 8), 0, 0);
#else
  const uint4* s = reinterpret_cast<const uint4*>(g);
  uint4* d = reinterpret_cast<uint4*>(l);
  d[0] = s[0]; d[1] = s[1];
#endif
}

__device__ __forceinline__ void wait_async_copies() {
#if HAVE_ASYNC_LDS
  __builtin_amdgcn_s_wait_asynccnt(0);
#endif
}

// ---------------------------------------------------------------------------
// fp32 -> fp16 conversion, 4 elements/thread
// ---------------------------------------------------------------------------
__global__ __launch_bounds__(256) void cvt_f32_to_f16_x4(
    const float* __restrict__ src, _Float16* __restrict__ dst, int n4) {
  int i = blockIdx.x * blockDim.x + threadIdx.x;
  if (i < n4) {
    float4 f = reinterpret_cast<const float4*>(src)[i];
    v4h h = {(_Float16)f.x, (_Float16)f.y, (_Float16)f.z, (_Float16)f.w};
    reinterpret_cast<v4h*>(dst)[i] = h;
  }
}

// ---------------------------------------------------------------------------
// Generic WMMA GEMM: C[M=4096, N=1024] = A[4096,1024] * W[N,K]^T + bias
// MODE 0: store f16 row-major [row][col]          (Q, K projections)
// MODE 1: store f16 transposed Vt[b][e][n]        (V projection)
// MODE 2: store f32 row-major [row][col]          (output projection -> d_out)
// Block: 256 threads = 8 waves, 128x64 output tile, waves 4x2, 32x32 each.
// Double-buffered LDS; next k-slab staged (async if available) during compute.
// ---------------------------------------------------------------------------
template<int MODE>
__global__ __launch_bounds__(256) void gemm128x64_wmma(
    const _Float16* __restrict__ A, const _Float16* __restrict__ W,
    const float* __restrict__ bias, void* __restrict__ Out) {
  constexpr int KD = 1024;
  constexpr int LS = 40;                       // LDS row stride in halves (80B)
  __shared__ _Float16 As[2][128 * LS];
  __shared__ _Float16 Bs[2][64 * LS];

  const int tid  = threadIdx.x;
  const int lane = tid & 31;
  const int wid  = tid >> 5;
  const int wm   = (wid >> 1) * 32;            // wave row offset in block tile
  const int wn   = (wid & 1) * 32;             // wave col offset
  const int half = lane >> 4;
  const int l15  = lane & 15;
  const int m0   = blockIdx.y * 128;
  const int n0   = blockIdx.x * 64;

  const int arow = tid >> 1;                   // 0..127
  const int ah   = (tid & 1) * 16;             // 0 or 16 halves

  const _Float16* gArow = A + (size_t)(m0 + arow) * KD + ah;
  const _Float16* gBrow = W + (size_t)(n0 + arow) * KD + ah;

  // Prologue: stage k-slab 0 into buffer 0.
  copy32_to_lds(gArow, &As[0][arow * LS + ah]);
  if (tid < 128) copy32_to_lds(gBrow, &Bs[0][arow * LS + ah]);
  wait_async_copies();
  __syncthreads();

  CFrag c[2][2] = {};

  for (int k0 = 0; k0 < KD; k0 += 32) {
    const int cur = (k0 >> 5) & 1;
    // Stage next k-slab into the other buffer while we compute this one.
    if (k0 + 32 < KD) {
      copy32_to_lds(gArow + k0 + 32, &As[cur ^ 1][arow * LS + ah]);
      if (tid < 128) copy32_to_lds(gBrow + k0 + 32, &Bs[cur ^ 1][arow * LS + ah]);
    }

    AFrag af[2], bf[2];
#pragma unroll
    for (int mt = 0; mt < 2; ++mt) {
      const _Float16* p = &As[cur][(wm + mt * 16 + l15) * LS];
      af[mt].u[0] = *reinterpret_cast<const uint4*>(p + half * 16);       // K 0-7 / 16-23
      af[mt].u[1] = *reinterpret_cast<const uint4*>(p + half * 16 + 8);   // K 8-15 / 24-31
    }
#pragma unroll
    for (int nt = 0; nt < 2; ++nt) {
      const _Float16* p = &Bs[cur][(wn + nt * 16 + l15) * LS] + half * 16; // K 0-15 / 16-31
      bf[nt].u[0] = *reinterpret_cast<const uint4*>(p);
      bf[nt].u[1] = *reinterpret_cast<const uint4*>(p + 8);
    }
#pragma unroll
    for (int mt = 0; mt < 2; ++mt)
#pragma unroll
      for (int nt = 0; nt < 2; ++nt)
        c[mt][nt].v = WMMA16x16x32(af[mt].v, bf[nt].v, c[mt][nt].v);

    if (k0 + 32 < KD) wait_async_copies();
    __syncthreads();
  }

  // Epilogue: D layout = VGPR r -> row r + 8*half, col = lane&15.
#pragma unroll
  for (int mt = 0; mt < 2; ++mt) {
    const int rowb = m0 + wm + mt * 16 + half * 8;
#pragma unroll
    for (int nt = 0; nt < 2; ++nt) {
      const int col = n0 + wn + nt * 16 + l15;
      const float bj = bias[col];
#pragma unroll
      for (int r = 0; r < 8; ++r) {
        const float v = c[mt][nt].f[r] + bj;
        const int gr = rowb + r;                       // global row = b*2048 + n
        if (MODE == 0) {
          reinterpret_cast<_Float16*>(Out)[(size_t)gr * 1024 + col] = (_Float16)v;
        } else if (MODE == 1) {
          const int bi = gr >> 11, ni = gr & 2047;
          reinterpret_cast<_Float16*>(Out)[(size_t)bi * (1024 * 2048)
                                           + (size_t)col * 2048 + ni] = (_Float16)v;
        } else {
          reinterpret_cast<float*>(Out)[(size_t)gr * 1024 + col] = v;
        }
      }
    }
  }
}

// ---------------------------------------------------------------------------
// Pass 1: per-key-column softmax stats.
//   S[n,m] = 0.25 * <q_n, k_m>, masked -inf where n > m.
//   cmax[m] = max_n S[n,m];  csum[m] = sum_n exp(S[n,m] - cmax[m]).
// One wave per (bh, 16-wide m tile). D-frag lanes hold fixed column m, so the
// over-n reduction is 8 VGPR slices + a shfl_xor(16) across the row halves.
// ---------------------------------------------------------------------------
__global__ __launch_bounds__(32) void attn_colstats(
    const _Float16* __restrict__ Q, const _Float16* __restrict__ Kmat,
    float* __restrict__ cmax, float* __restrict__ csum) {
  const int mt = blockIdx.x;                   // 0..127
  const int bh = blockIdx.y;                   // 0..31
  const int b = bh >> 4, h = bh & 15;
  const int lane = threadIdx.x;
  const int half = lane >> 4, l15 = lane & 15;
  const int m0 = mt * 16;
  const int colm = m0 + l15;

  const _Float16* Krow = Kmat + (size_t)(b * 2048 + colm) * 1024 + h * 64;
  AFrag kb0, kb1;                               // B frags of K^T, kept live
  kb0.u[0] = *reinterpret_cast<const uint4*>(Krow + half * 16);
  kb0.u[1] = *reinterpret_cast<const uint4*>(Krow + half * 16 + 8);
  kb1.u[0] = *reinterpret_cast<const uint4*>(Krow + 32 + half * 16);
  kb1.u[1] = *reinterpret_cast<const uint4*>(Krow + 32 + half * 16 + 8);

  float runmax = -INFINITY, runsum = 0.f;
  for (int nt = 0; nt <= mt; ++nt) {
    const int n0 = nt * 16;
    const _Float16* Qrow = Q + (size_t)(b * 2048 + n0 + l15) * 1024 + h * 64;
    if (nt < mt) __builtin_prefetch(Qrow + 16 * 1024, 0, 3);   // next n-tile row
    AFrag a0, a1;
    a0.u[0] = *reinterpret_cast<const uint4*>(Qrow + half * 16);
    a0.u[1] = *reinterpret_cast<const uint4*>(Qrow + half * 16 + 8);
    a1.u[0] = *reinterpret_cast<const uint4*>(Qrow + 32 + half * 16);
    a1.u[1] = *reinterpret_cast<const uint4*>(Qrow + 32 + half * 16 + 8);
    CFrag s = {};
    s.v = WMMA16x16x32(a0.v, kb0.v, s.v);
    s.v = WMMA16x16x32(a1.v, kb1.v, s.v);

    float sv[8], tmax = -INFINITY;
#pragma unroll
    for (int r = 0; r < 8; ++r) {
      const int n = n0 + half * 8 + r;
      sv[r] = (n <= colm) ? s.f[r] * 0.25f : -INFINITY;
      tmax = fmaxf(tmax, sv[r]);
    }
    tmax = fmaxf(tmax, __shfl_xor(tmax, 16, 32));
    const float newmax = fmaxf(runmax, tmax);
    float tsum = 0.f;
#pragma unroll
    for (int r = 0; r < 8; ++r) tsum += __expf(sv[r] - newmax);
    tsum += __shfl_xor(tsum, 16, 32);
    runsum = runsum * __expf(runmax - newmax) + tsum;
    runmax = newmax;
  }
  if (lane < 16) {
    cmax[(size_t)bh * 2048 + colm] = runmax;
    csum[(size_t)bh * 2048 + colm] = runsum;
  }
}

// ---------------------------------------------------------------------------
// Pass 2: O[:, n] = sum_m exp(S[n,m]-cmax[m])/csum[m] * v_m.
// One wave per (bh, 16-wide n tile). Recompute S tiles via WMMA, build P in
// LDS (D-frag -> A-frag transpose), then O += P x V with V read from the
// pre-transposed Vt so B-fragment loads are contiguous.
// ---------------------------------------------------------------------------
__global__ __launch_bounds__(32) void attn_apply(
    const _Float16* __restrict__ Q, const _Float16* __restrict__ Kmat,
    const _Float16* __restrict__ Vt, const float* __restrict__ cmax,
    const float* __restrict__ csum, _Float16* __restrict__ AttnOut) {
  __shared__ _Float16 Ps[16 * 40];             // 16 n-rows x 32 m-cols, stride 40
  const int ntile = blockIdx.x;
  const int bh = blockIdx.y;
  const int b = bh >> 4, h = bh & 15;
  const int lane = threadIdx.x;
  const int half = lane >> 4, l15 = lane & 15;
  const int n0 = ntile * 16;

  const _Float16* Qrow = Q + (size_t)(b * 2048 + n0 + l15) * 1024 + h * 64;
  AFrag qa0, qa1;                               // A frags of Q, fixed per block
  qa0.u[0] = *reinterpret_cast<const uint4*>(Qrow + half * 16);
  qa0.u[1] = *reinterpret_cast<const uint4*>(Qrow + half * 16 + 8);
  qa1.u[0] = *reinterpret_cast<const uint4*>(Qrow + 32 + half * 16);
  qa1.u[1] = *reinterpret_cast<const uint4*>(Qrow + 32 + half * 16 + 8);

  CFrag o[4] = {};
  for (int mc = (n0 & ~31); mc < 2048; mc += 32) {
#pragma unroll
    for (int t = 0; t < 2; ++t) {
      const int colm = mc + t * 16 + l15;
      const _Float16* Krow = Kmat + (size_t)(b * 2048 + colm) * 1024 + h * 64;
      if (mc + 32 < 2048) __builtin_prefetch(Krow + 32 * 1024, 0, 3);  // next m-chunk
      AFrag kb0, kb1;
      kb0.u[0] = *reinterpret_cast<const uint4*>(Krow + half * 16);
      kb0.u[1] = *reinterpret_cast<const uint4*>(Krow + half * 16 + 8);
      kb1.u[0] = *reinterpret_cast<const uint4*>(Krow + 32 + half * 16);
      kb1.u[1] = *reinterpret_cast<const uint4*>(Krow + 32 + half * 16 + 8);
      CFrag s = {};
      s.v = WMMA16x16x32(qa0.v, kb0.v, s.v);
      s.v = WMMA16x16x32(qa1.v, kb1.v, s.v);
      const float cm  = cmax[(size_t)bh * 2048 + colm];
      const float inv = 1.0f / csum[(size_t)bh * 2048 + colm];
#pragma unroll
      for (int r = 0; r < 8; ++r) {
        const int n = n0 + half * 8 + r;
        const float w = (n <= colm) ? __expf(s.f[r] * 0.25f - cm) * inv : 0.f;
        Ps[(half * 8 + r) * 40 + t * 16 + l15] = (_Float16)w;
      }
    }
    __syncthreads();
    AFrag pa;                                   // P as 16x32 A-fragment
    pa.u[0] = *reinterpret_cast<const uint4*>(&Ps[l15 * 40 + half * 16]);
    pa.u[1] = *reinterpret_cast<const uint4*>(&Ps[l15 * 40 + half * 16 + 8]);
#pragma unroll
    for (int jt = 0; jt < 4; ++jt) {
      const int e = h * 64 + jt * 16 + l15;
      const _Float16* Vrow = Vt + (size_t)b * (1024 * 2048) + (size_t)e * 2048
                             + mc + half * 16;
      AFrag vb;
      vb.u[0] = *reinterpret_cast<const uint4*>(Vrow);
      vb.u[1] = *reinterpret_cast<const uint4*>(Vrow + 8);
      o[jt].v = WMMA16x16x32(pa.v, vb.v, o[jt].v);
    }
    __syncthreads();
  }

#pragma unroll
  for (int jt = 0; jt < 4; ++jt) {
    const int e = h * 64 + jt * 16 + l15;
#pragma unroll
    for (int r = 0; r < 8; ++r) {
      const int n = n0 + half * 8 + r;
      AttnOut[(size_t)(b * 2048 + n) * 1024 + e] = (_Float16)o[jt].f[r];
    }
  }
}

// ---------------------------------------------------------------------------
extern "C" void kernel_launch(void* const* d_in, const int* in_sizes, int n_in,
                              void* d_out, int out_size, void* d_ws, size_t ws_size,
                              hipStream_t stream) {
  const float* x  = (const float*)d_in[0];
  const float* Wq = (const float*)d_in[1];
  const float* bq = (const float*)d_in[2];
  const float* Wk = (const float*)d_in[3];
  const float* bk = (const float*)d_in[4];
  const float* Wv = (const float*)d_in[5];
  const float* bv = (const float*)d_in[6];
  const float* Wp = (const float*)d_in[7];
  const float* bp = (const float*)d_in[8];

  char* w = (char*)d_ws;
  auto alloc = [&](size_t bytes) -> char* {
    char* p = w; w += (bytes + 255) & ~(size_t)255; return p;
  };
  _Float16* Xh   = (_Float16*)alloc(4096ull * 1024 * 2);   // x in f16
  _Float16* Wqh  = (_Float16*)alloc(1024ull * 1024 * 2);
  _Float16* Wkh  = (_Float16*)alloc(1024ull * 1024 * 2);
  _Float16* Wvh  = (_Float16*)alloc(1024ull * 1024 * 2);
  _Float16* Wph  = (_Float16*)alloc(1024ull * 1024 * 2);
  _Float16* Qm   = (_Float16*)alloc(4096ull * 1024 * 2);   // Q rows (b,n) x e
  _Float16* Km   = (_Float16*)alloc(4096ull * 1024 * 2);   // K rows (b,n) x e
  _Float16* Vt   = (_Float16*)alloc(4096ull * 1024 * 2);   // Vt[b][e][n]
  _Float16* At   = (_Float16*)alloc(4096ull * 1024 * 2);   // attention output
  float*    cmax = (float*)alloc(32ull * 2048 * 4);
  float*    csum = (float*)alloc(32ull * 2048 * 4);

  // f32 -> f16 conversions
  {
    int n4 = (2 * 2048 * 1024) / 4;
    cvt_f32_to_f16_x4<<<(n4 + 255) / 256, 256, 0, stream>>>(x, Xh, n4);
    int w4 = (1024 * 1024) / 4;
    cvt_f32_to_f16_x4<<<(w4 + 255) / 256, 256, 0, stream>>>(Wq, Wqh, w4);
    cvt_f32_to_f16_x4<<<(w4 + 255) / 256, 256, 0, stream>>>(Wk, Wkh, w4);
    cvt_f32_to_f16_x4<<<(w4 + 255) / 256, 256, 0, stream>>>(Wv, Wvh, w4);
    cvt_f32_to_f16_x4<<<(w4 + 255) / 256, 256, 0, stream>>>(Wp, Wph, w4);
  }

  dim3 gg(1024 / 64, 4096 / 128);   // (16, 32)
  gemm128x64_wmma<0><<<gg, 256, 0, stream>>>(Xh, Wqh, bq, (void*)Qm);
  gemm128x64_wmma<0><<<gg, 256, 0, stream>>>(Xh, Wkh, bk, (void*)Km);
  gemm128x64_wmma<1><<<gg, 256, 0, stream>>>(Xh, Wvh, bv, (void*)Vt);

  attn_colstats<<<dim3(2048 / 16, 32), 32, 0, stream>>>(Qm, Km, cmax, csum);
  attn_apply<<<dim3(2048 / 16, 32), 32, 0, stream>>>(Qm, Km, Vt, cmax, csum, At);

  gemm128x64_wmma<2><<<gg, 256, 0, stream>>>(At, Wph, bp, d_out);
}
